// RecallLayer_51548197487002
// MI455X (gfx1250) — matrix operand order repaired and checked by reading, not compiled, not run on hardware
//
#include <hip/hip_runtime.h>

// ---------------------------------------------------------------------------
// MI455X (gfx1250, wave32) implementation of RecallLayer.
// All dense algebra is mapped onto v_wmma_f32_16x16x32_bf16 (16x16x32 tiles),
// fp32 accumulate.  Attention is flash-style: rk tile staged in LDS, score
// tiles distributed one-key-block-per-wave (no redundant score GEMM), masked
// in-register, reshaped through LDS into the A-fragment layout.
// ---------------------------------------------------------------------------

constexpr int Bc   = 2;
constexpr int Tc   = 4096;
constexpr int Dc   = 1024;
constexpr int EIc  = 2048;
constexpr int BT   = Bc * Tc;          // 8192 tokens

typedef __bf16 bf16x16 __attribute__((ext_vector_type(16)));
typedef float  f32x8   __attribute__((ext_vector_type(8)));

#define DEVFN __device__ __forceinline__

DEVFN unsigned short f2bf(float f) {
  unsigned int u = __float_as_uint(f);
  u += 0x7FFFu + ((u >> 16) & 1u);           // round-to-nearest-even
  return (unsigned short)(u >> 16);
}

DEVFN float sigmoidf_(float x) { return 1.f / (1.f + __expf(-x)); }
DEVFN float siluf_(float x)    { return x * sigmoidf_(x); }

// A-fragment (16x32 bf16, M x K): lane&15 = row M; lanes<16 hold K 0..7,16..23,
// lanes>=16 hold K 8..15,24..31  -> two 16B loads per lane.
DEVFN bf16x16 load_a_frag(const unsigned short* p0, int lda) {
  int lane = threadIdx.x & 31;
  const unsigned short* p = p0 + (size_t)(lane & 15) * lda + ((lane >> 4) << 3);
  union { bf16x16 v; uint4 q[2]; } u;
  u.q[0] = *(const uint4*)(p);
  u.q[1] = *(const uint4*)(p + 16);
  return u.v;
}

// B-fragment (32x16 bf16, K x N) sourced from a row-major (N,K) "weight":
// lane&15 = column N; lanes<16 hold K 0..7,16..23, lanes>=16 hold K 8..15,24..31.
DEVFN bf16x16 load_b_frag(const unsigned short* p0, int ldb) {
  int lane = threadIdx.x & 31;
  const unsigned short* p = p0 + (size_t)(lane & 15) * ldb + ((lane >> 4) << 3);
  union { bf16x16 v; uint4 q[2]; } u;
  u.q[0] = *(const uint4*)(p);
  u.q[1] = *(const uint4*)(p + 16);
  return u.v;
}

DEVFN f32x8 wmma_bf16(bf16x16 a, bf16x16 b, f32x8 c) {
  return __builtin_amdgcn_wmma_f32_16x16x32_bf16(false, a, false, b,
                                                 (short)0, c, false, false);
}

// ---------------------------------------------------------------------------
// Generic C[M,N] = A[M,K](bf16) * W[N,K](bf16)^T  with epilogues.
// 128 threads = 4 waves; wave w owns rows (blockIdx.y*4+w)*16..+15,
// NT 16-wide column tiles.  M % 64 == 0, N % (16*NT) == 0, K % 32 == 0.
// EPI: 0 = plain   1 = softplus(acc+bias[n])
//      2 = add0+add1+alpha*acc               (final residual projection)
//      3 = acc + exp(T*log_gamma)*Wold       (W_new)
// ---------------------------------------------------------------------------
template<int NT, int EPI>
__global__ __launch_bounds__(128)
void gemm_bf16(const unsigned short* __restrict__ A, int lda,
               const unsigned short* __restrict__ Wt, int ldw,
               float* __restrict__ C, int ldc, int Kd,
               const float* __restrict__ bias,
               const float* __restrict__ add0,
               const float* __restrict__ add1,
               float alpha,
               const float* __restrict__ Wold,
               int Tlen,
               const float* __restrict__ decay_p) {
  const int wave = threadIdx.x >> 5;
  const int m0 = (blockIdx.y * 4 + wave) * 16;
  const int n0 = blockIdx.x * (16 * NT);
  const unsigned short* Ap = A + (size_t)m0 * lda;

  f32x8 acc[NT];
#pragma unroll
  for (int i = 0; i < NT; ++i)
#pragma unroll
    for (int j = 0; j < 8; ++j) acc[i][j] = 0.f;

  for (int k0 = 0; k0 < Kd; k0 += 32) {
    __builtin_prefetch((const void*)(Ap + k0 + 128), 0, 1);   // global_prefetch_b8
    bf16x16 a = load_a_frag(Ap + k0, lda);
#pragma unroll
    for (int i = 0; i < NT; ++i) {
      bf16x16 b = load_b_frag(Wt + (size_t)(n0 + 16 * i) * ldw + k0, ldw);
      acc[i] = wmma_bf16(a, b, acc[i]);
    }
  }

  const int lane = threadIdx.x & 31;
  const int col0 = lane & 15;
  const int rb = (lane >> 4) << 3;
  float wdk = 0.f;
  if (EPI == 3) {
    float lg = logf(sigmoidf_(decay_p[0]));
    wdk = __expf((float)Tlen * lg);
  }
#pragma unroll
  for (int i = 0; i < NT; ++i) {
    int col = n0 + 16 * i + col0;
#pragma unroll
    for (int j = 0; j < 8; ++j) {
      int row = m0 + rb + j;
      size_t idx = (size_t)row * ldc + col;
      float v = acc[i][j];
      float o;
      if (EPI == 0)      o = v;
      else if (EPI == 1) { float t = v + bias[col];
                           o = (t > 20.f) ? t : log1pf(__expf(t)); }
      else if (EPI == 2) o = add0[idx] + add1[idx] + alpha * v;
      else               o = v + wdk * Wold[idx];
      C[idx] = o;
    }
  }
}

// ------------------------- elementwise / prep kernels ----------------------

__global__ void f2bf_kernel(const float* __restrict__ in,
                            unsigned short* __restrict__ out, size_t n) {
  size_t i = (size_t)blockIdx.x * blockDim.x + threadIdx.x;
  if (i < n) out[i] = f2bf(in[i]);
}

__global__ void rmsnorm_kernel(const float* __restrict__ x,
                               const float* __restrict__ nw,
                               unsigned short* __restrict__ xn) {
  int t = blockIdx.x;                               // token
  const float* xr = x + (size_t)t * Dc;
  __shared__ float s[256];
  float acc = 0.f;
  for (int i = threadIdx.x; i < Dc; i += 256) { float v = xr[i]; acc += v * v; }
  s[threadIdx.x] = acc; __syncthreads();
  for (int st = 128; st > 0; st >>= 1) {
    if ((int)threadIdx.x < st) s[threadIdx.x] += s[threadIdx.x + st];
    __syncthreads();
  }
  float r = rsqrtf(s[0] * (1.f / Dc) + 1e-6f);
  for (int i = threadIdx.x; i < Dc; i += 256)
    xn[(size_t)t * Dc + i] = f2bf(xr[i] * r * nw[i]);
}

// causal depthwise conv (K=4) + bias + SiLU; xp = xz[:, :EI] (row stride 2*EI)
__global__ void conv_silu_kernel(const float* __restrict__ xz,
                                 const float* __restrict__ cw,
                                 const float* __restrict__ cb,
                                 float* __restrict__ xc,
                                 unsigned short* __restrict__ xcb) {
  size_t idx = (size_t)blockIdx.x * 256 + threadIdx.x;   // over BT*EI
  if (idx >= (size_t)BT * EIc) return;
  int e = (int)(idx & (EIc - 1));
  size_t bt = idx >> 11;                                  // EIc = 2^11
  int t = (int)(bt & (Tc - 1));
  size_t btb = bt - t;                                    // batch start row
  float s = cb[e];
#pragma unroll
  for (int k = 0; k < 4; ++k) {
    int tt = t - 3 + k;
    if (tt >= 0) s += cw[e * 4 + k] * xz[(btb + tt) * (size_t)(2 * EIc) + e];
  }
  float o = siluf_(s);
  xc[idx] = o;
  xcb[idx] = f2bf(o);
}

__global__ void split_dbl_kernel(const float* __restrict__ dbl,
                                 unsigned short* __restrict__ dtin) {
  size_t i = (size_t)blockIdx.x * 256 + threadIdx.x;      // over BT*64
  if (i >= (size_t)BT * 64) return;
  int r = (int)(i & 63);
  size_t bt = i >> 6;
  dtin[i] = f2bf(dbl[bt * 96 + r]);
}

// selective scan: one thread per (b,e); B_t/C_t staged in LDS per step;
// gating + D skip + SiLU(z) fused, writes bf16 GEMM operand.
__global__ void scan_kernel(const float* __restrict__ dt,
                            const float* __restrict__ xc,
                            const float* __restrict__ dbl,
                            const float* __restrict__ A_log,
                            const float* __restrict__ Dp,
                            const float* __restrict__ xz,
                            unsigned short* __restrict__ yg) {
  int b = blockIdx.x >> 3;                       // EIc/256 = 8 blocks per batch
  int e = ((blockIdx.x & 7) << 8) + threadIdx.x;
  float A[16], h[16];
#pragma unroll
  for (int n = 0; n < 16; ++n) { A[n] = -__expf(A_log[e * 16 + n]); h[n] = 0.f; }
  float Dpe = Dp[e];
  __shared__ float sB[16], sC[16];
  for (int t = 0; t < Tc; ++t) {
    size_t bt = (size_t)b * Tc + t;
    if (threadIdx.x < 32) {
      float v = dbl[bt * 96 + 64 + threadIdx.x];
      if (threadIdx.x < 16) sB[threadIdx.x] = v; else sC[threadIdx.x - 16] = v;
    }
    __syncthreads();
    float dtv = dt[bt * EIc + e];
    float u   = xc[bt * EIc + e];
    float du  = dtv * u;
    float y = 0.f;
#pragma unroll
    for (int n = 0; n < 16; ++n) {
      h[n] = __expf(dtv * A[n]) * h[n] + du * sB[n];
      y += h[n] * sC[n];
    }
    float zv = xz[bt * (size_t)(2 * EIc) + EIc + e];
    yg[bt * EIc + e] = f2bf((y + u * Dpe) * siluf_(zv));
    __syncthreads();
  }
}

// rk = bf16(out); wk = bf16(out shifted right by one token, zero at t==0)
__global__ void rkwk_kernel(const float* __restrict__ mo,
                            unsigned short* __restrict__ rkb,
                            unsigned short* __restrict__ wkb) {
  size_t i = (size_t)blockIdx.x * 256 + threadIdx.x;      // over BT*D
  if (i >= (size_t)BT * Dc) return;
  int t = (int)((i >> 10) & (Tc - 1));
  rkb[i] = f2bf(mo[i]);
  wkb[i] = (t == 0) ? (unsigned short)0 : f2bf(mo[i - Dc]);
}

// Build vT[b,d,t], gamma-scaled vT, and wkT[b,e,t] (bf16) for attention / W_new
__global__ void transp_kernel(const float* __restrict__ v,
                              const float* __restrict__ mo,
                              unsigned short* __restrict__ vT,
                              unsigned short* __restrict__ vsT,
                              unsigned short* __restrict__ wkT,
                              const float* __restrict__ decay_p) {
  size_t i = (size_t)blockIdx.x * 256 + threadIdx.x;      // over B*D*T (t fastest)
  if (i >= (size_t)Bc * Dc * Tc) return;
  int t = (int)(i & (Tc - 1));
  size_t bd = i >> 12;
  int d = (int)(bd & (Dc - 1));
  int b = (int)(bd >> 10);
  float lg = logf(sigmoidf_(decay_p[0]));
  size_t src = ((size_t)b * Tc + t) * Dc + d;
  float vv = v[src];
  vT[i]  = f2bf(vv);
  vsT[i] = f2bf(vv * __expf((float)(Tc - 1 - t) * lg));
  wkT[i] = (t == 0) ? (unsigned short)0 : f2bf(mo[src - Dc]);
}

// ---------------------------------------------------------------------------
// Flash-style decay attention.
// Block = 8 waves, one 16-query tile.  rk tile (16x1024 bf16, 32KB) staged in
// LDS once.  Key blocks processed in groups of 8: wave w owns key block
// g*8+w, computes its 16x32 masked score tile (WMMA over K=1024, A-frags from
// LDS), writes it to its LDS slot; after a barrier every wave consumes all
// valid score tiles for its own 128 output columns (A-frags from LDS, B-frags
// from the pre-transposed vT).  No redundant score GEMM, no spill pressure
// (#pragma unroll 1 keeps one k-step of fragments live).
// ---------------------------------------------------------------------------
__global__ __launch_bounds__(256)
void attn_kernel(const unsigned short* __restrict__ rk,
                 const unsigned short* __restrict__ wk,
                 const unsigned short* __restrict__ vT,
                 float* __restrict__ reads,
                 const float* __restrict__ decay_p) {
  __shared__ unsigned short srk[16 * Dc];       // 32 KB rk tile
  __shared__ unsigned short sS[8][16 * 32];     // 8 KB masked score tiles
  const int wave = threadIdx.x >> 5, lane = threadIdx.x & 31;
  const int qt = blockIdx.x;                    // global 16-query tile
  const int b  = (qt * 16) / Tc;
  const int q0 = qt * 16 - b * Tc;              // batch-local query base
  const float lg = logf(sigmoidf_(decay_p[0]));
  const unsigned short* rkb = rk + ((size_t)b * Tc + q0) * Dc;
  const unsigned short* wkb = wk + (size_t)b * Tc * Dc;
  const unsigned short* vTb = vT + (size_t)b * Dc * Tc;
  const int n0 = wave * 128;
  const int col0 = lane & 15, rb = (lane >> 4) << 3;

  // cooperative stage of the rk tile into LDS (2048 x 16B)
  {
    const uint4* src = (const uint4*)rkb;
    uint4* dst = (uint4*)srk;
    for (int i = threadIdx.x; i < 2048; i += 256) dst[i] = src[i];
  }
  asm volatile("s_wait_dscnt 0x0" ::: "memory");
  __syncthreads();

  f32x8 acc[8];
#pragma unroll
  for (int i = 0; i < 8; ++i)
#pragma unroll
    for (int j = 0; j < 8; ++j) acc[i][j] = 0.f;

  const int kbmax = (q0 + 14) / 32;             // last key block with any j < i
  for (int g = 0; g <= kbmax / 8; ++g) {
    const int kb = g * 8 + wave;
    if (kb <= kbmax) {                          // this wave produces one tile
      const int j0 = kb * 32;
      f32x8 s0, s1;
#pragma unroll
      for (int j = 0; j < 8; ++j) { s0[j] = 0.f; s1[j] = 0.f; }
#pragma unroll 1
      for (int k0 = 0; k0 < Dc; k0 += 32) {
        bf16x16 a  = load_a_frag(srk + k0, Dc);               // ds_load_b128
        bf16x16 b0 = load_b_frag(wkb + (size_t)j0 * Dc + k0, Dc);
        bf16x16 b1 = load_b_frag(wkb + (size_t)(j0 + 16) * Dc + k0, Dc);
        s0 = wmma_bf16(a, b0, s0);
        s1 = wmma_bf16(a, b1, s1);
      }
      unsigned short* sw = sS[wave];
#pragma unroll
      for (int j = 0; j < 8; ++j) {             // decay mask in C layout
        int qi = q0 + rb + j;
        int k1 = j0 + col0, k2 = j0 + 16 + col0;
        float m1 = (qi > k1) ? __expf((float)(qi - 1 - k1) * lg) : 0.f;
        float m2 = (qi > k2) ? __expf((float)(qi - 1 - k2) * lg) : 0.f;
        sw[(rb + j) * 32 + col0]      = f2bf(s0[j] * m1);
        sw[(rb + j) * 32 + 16 + col0] = f2bf(s1[j] * m2);
      }
    }
    __syncthreads();
    int kend = kbmax + 1 - g * 8; if (kend > 8) kend = 8;     // uniform
#pragma unroll 1
    for (int kk = 0; kk < kend; ++kk) {
      const int j0 = (g * 8 + kk) * 32;
      bf16x16 a2 = load_a_frag(sS[kk], 32);     // masked scores, A layout
#pragma unroll
      for (int i = 0; i < 8; ++i) {
        bf16x16 b2 = load_b_frag(vTb + (size_t)(n0 + 16 * i) * Tc + j0, Tc);
        acc[i] = wmma_bf16(a2, b2, acc[i]);
      }
    }
    __syncthreads();
  }

  float* Rb = reads + ((size_t)b * Tc + q0) * Dc;
#pragma unroll
  for (int i = 0; i < 8; ++i)
#pragma unroll
    for (int j = 0; j < 8; ++j)
      Rb[(size_t)(rb + j) * Dc + n0 + 16 * i + col0] = acc[i][j];
}

// reads_total = attn_reads + carry * gamma^t  (bf16 operand for proj_read GEMM)
__global__ void combine_kernel(const float* __restrict__ reads,
                               const float* __restrict__ carry,
                               unsigned short* __restrict__ rt,
                               const float* __restrict__ decay_p) {
  size_t i = (size_t)blockIdx.x * 256 + threadIdx.x;      // over BT*D
  if (i >= (size_t)BT * Dc) return;
  int t = (int)((i >> 10) & (Tc - 1));
  float lg = logf(sigmoidf_(decay_p[0]));
  rt[i] = f2bf(reads[i] + carry[i] * __expf((float)t * lg));
}

// ---------------------------------------------------------------------------

extern "C" void kernel_launch(void* const* d_in, const int* in_sizes, int n_in,
                              void* d_out, int out_size, void* d_ws, size_t ws_size,
                              hipStream_t stream) {
  const float* x     = (const float*)d_in[0];
  const float* Wmem  = (const float*)d_in[1];
  const float* normw = (const float*)d_in[2];
  const float* inprj = (const float*)d_in[3];
  const float* convw = (const float*)d_in[4];
  const float* convb = (const float*)d_in[5];
  const float* xproj = (const float*)d_in[6];
  const float* dtw   = (const float*)d_in[7];
  const float* dtb   = (const float*)d_in[8];
  const float* Alog  = (const float*)d_in[9];
  const float* Dpar  = (const float*)d_in[10];
  const float* outp  = (const float*)d_in[11];
  const float* pwr   = (const float*)d_in[12];
  const float* prd   = (const float*)d_in[13];
  const float* decay = (const float*)d_in[14];
  float* out0 = (float*)d_out;                       // x + out   [BT*D]
  float* outW = out0 + (size_t)BT * Dc;              // W_new     [B*D*D]

  // workspace allocator
  char* base = (char*)d_ws; size_t off = 0;
  auto alloc = [&](size_t bytes) { void* p = base + off;
                                   off = (off + bytes + 255) & ~(size_t)255;
                                   return p; };
  unsigned short* xn_bf   = (unsigned short*)alloc((size_t)BT * Dc * 2);
  unsigned short* w_in_bf = (unsigned short*)alloc((size_t)2 * EIc * Dc * 2);
  unsigned short* w_xp_bf = (unsigned short*)alloc((size_t)96 * EIc * 2);
  unsigned short* w_dt_bf = (unsigned short*)alloc((size_t)EIc * 64 * 2);
  unsigned short* w_op_bf = (unsigned short*)alloc((size_t)Dc * EIc * 2);
  unsigned short* w_pw_bf = (unsigned short*)alloc((size_t)Dc * Dc * 2);
  unsigned short* w_pr_bf = (unsigned short*)alloc((size_t)Dc * Dc * 2);
  unsigned short* Wb_bf   = (unsigned short*)alloc((size_t)Bc * Dc * Dc * 2);
  float* xz    = (float*)alloc((size_t)BT * 2 * EIc * 4);
  float* xc    = (float*)alloc((size_t)BT * EIc * 4);
  unsigned short* xc_bf   = (unsigned short*)alloc((size_t)BT * EIc * 2);
  float* dbl   = (float*)alloc((size_t)BT * 96 * 4);
  unsigned short* dtin_bf = (unsigned short*)alloc((size_t)BT * 64 * 2);
  float* dt    = (float*)alloc((size_t)BT * EIc * 4);
  unsigned short* yg_bf   = (unsigned short*)alloc((size_t)BT * EIc * 2);
  float* mo    = (float*)alloc((size_t)BT * Dc * 4);
  unsigned short* rk_bf   = (unsigned short*)alloc((size_t)BT * Dc * 2);
  unsigned short* wk_bf   = (unsigned short*)alloc((size_t)BT * Dc * 2);
  float* v     = (float*)alloc((size_t)BT * Dc * 4);
  unsigned short* vT_bf   = (unsigned short*)alloc((size_t)Bc * Dc * Tc * 2);
  unsigned short* vsT_bf  = (unsigned short*)alloc((size_t)Bc * Dc * Tc * 2);
  unsigned short* wkT_bf  = (unsigned short*)alloc((size_t)Bc * Dc * Tc * 2);
  float* reads = (float*)alloc((size_t)BT * Dc * 4);
  float* carry = (float*)alloc((size_t)BT * Dc * 4);
  unsigned short* rt_bf   = (unsigned short*)alloc((size_t)BT * Dc * 2);
  (void)ws_size; (void)n_in; (void)in_sizes; (void)out_size;

  auto cvt = [&](const float* src, unsigned short* dst, size_t n) {
    f2bf_kernel<<<dim3((unsigned)((n + 255) / 256)), 256, 0, stream>>>(src, dst, n);
  };
  // weight conversions (fp32 -> bf16)
  cvt(inprj, w_in_bf, (size_t)2 * EIc * Dc);
  cvt(xproj, w_xp_bf, (size_t)96 * EIc);
  cvt(dtw,   w_dt_bf, (size_t)EIc * 64);
  cvt(outp,  w_op_bf, (size_t)Dc * EIc);
  cvt(pwr,   w_pw_bf, (size_t)Dc * Dc);
  cvt(prd,   w_pr_bf, (size_t)Dc * Dc);
  cvt(Wmem,  Wb_bf,   (size_t)Bc * Dc * Dc);

  dim3 blk(128);
  // 1. RMSNorm -> bf16
  rmsnorm_kernel<<<BT, 256, 0, stream>>>(x, normw, xn_bf);
  // 2. xz = xn @ in_proj^T           [8192 x 4096, K=1024]
  gemm_bf16<4,0><<<dim3(2*EIc/64, BT/64), blk, 0, stream>>>(
      xn_bf, Dc, w_in_bf, Dc, xz, 2*EIc, Dc,
      nullptr, nullptr, nullptr, 0.f, nullptr, 0, nullptr);
  // 3. causal conv + SiLU
  conv_silu_kernel<<<(unsigned)(((size_t)BT*EIc+255)/256), 256, 0, stream>>>(
      xz, convw, convb, xc, xc_bf);
  // 4. dbl = xc @ x_proj^T           [8192 x 96, K=2048]
  gemm_bf16<2,0><<<dim3(96/32, BT/64), blk, 0, stream>>>(
      xc_bf, EIc, w_xp_bf, EIc, dbl, 96, EIc,
      nullptr, nullptr, nullptr, 0.f, nullptr, 0, nullptr);
  split_dbl_kernel<<<(unsigned)(((size_t)BT*64+255)/256), 256, 0, stream>>>(dbl, dtin_bf);
  // 5. dt = softplus(dbl[:, :64] @ dt_proj^T + b)   [8192 x 2048, K=64]
  gemm_bf16<4,1><<<dim3(EIc/64, BT/64), blk, 0, stream>>>(
      dtin_bf, 64, w_dt_bf, 64, dt, EIc, 64,
      dtb, nullptr, nullptr, 0.f, nullptr, 0, nullptr);
  // 6. selective scan (+gating) -> yg bf16
  scan_kernel<<<Bc * (EIc/256), 256, 0, stream>>>(dt, xc, dbl, Alog, Dpar, xz, yg_bf);
  // 7. mamba out = yg @ out_proj^T   [8192 x 1024, K=2048]
  gemm_bf16<4,0><<<dim3(Dc/64, BT/64), blk, 0, stream>>>(
      yg_bf, EIc, w_op_bf, EIc, mo, Dc, EIc,
      nullptr, nullptr, nullptr, 0.f, nullptr, 0, nullptr);
  // 8. rk / wk (shifted) bf16
  rkwk_kernel<<<(unsigned)(((size_t)BT*Dc+255)/256), 256, 0, stream>>>(mo, rk_bf, wk_bf);
  // 9. v = out @ proj_write^T        [8192 x 1024, K=1024]
  gemm_bf16<4,0><<<dim3(Dc/64, BT/64), blk, 0, stream>>>(
      rk_bf, Dc, w_pw_bf, Dc, v, Dc, Dc,
      nullptr, nullptr, nullptr, 0.f, nullptr, 0, nullptr);
  // 10. transposed bf16 operands for attention and W_new
  transp_kernel<<<(unsigned)(((size_t)Bc*Dc*Tc+255)/256), 256, 0, stream>>>(
      v, mo, vT_bf, vsT_bf, wkT_bf, decay);
  // 11. decay-masked attention -> reads
  attn_kernel<<<BT/16, 256, 0, stream>>>(rk_bf, wk_bf, vT_bf, reads, decay);
  // 12. carry = W[b] @ rk^T per batch [4096 x 1024, K=1024]
  for (int b = 0; b < Bc; ++b)
    gemm_bf16<4,0><<<dim3(Dc/64, Tc/64), blk, 0, stream>>>(
        rk_bf + (size_t)b*Tc*Dc, Dc, Wb_bf + (size_t)b*Dc*Dc, Dc,
        carry + (size_t)b*Tc*Dc, Dc, Dc,
        nullptr, nullptr, nullptr, 0.f, nullptr, 0, nullptr);
  // 13. reads_total = reads + carry*gamma^t  -> bf16
  combine_kernel<<<(unsigned)(((size_t)BT*Dc+255)/256), 256, 0, stream>>>(
      reads, carry, rt_bf, decay);
  // 14. out0 = x + mo + 0.03 * reads_total @ proj_read^T
  gemm_bf16<4,2><<<dim3(Dc/64, BT/64), blk, 0, stream>>>(
      rt_bf, Dc, w_pr_bf, Dc, out0, Dc, Dc,
      nullptr, x, mo, 0.03f, nullptr, 0, nullptr);
  // 15. W_new[b] = (gamma-scaled v)^T @ wk + gamma^T * W   [1024 x 1024, K=4096]
  for (int b = 0; b < Bc; ++b)
    gemm_bf16<4,3><<<dim3(Dc/64, Dc/64), blk, 0, stream>>>(
        vsT_bf + (size_t)b*Dc*Tc, Tc, wkT_bf + (size_t)b*Dc*Tc, Tc,
        outW + (size_t)b*Dc*Dc, Dc, Tc,
        nullptr, nullptr, nullptr, 0.f, Wmem + (size_t)b*Dc*Dc, Tc, decay);
}
